// LongformerClassifier_38285338477170
// MI455X (gfx1250) — compile-verified, matrix operand order checked by reading, not run on hardware
//
#include <hip/hip_runtime.h>
#include <hip/hip_bf16.h>
#include <stdint.h>

// ---------------- model constants ----------------
#define L_   12
#define D_   768
#define H_   12
#define DH_  64
#define FF_  3072
#define S_   2048
#define W_   128
#define C_   (S_/W_)
#define G_   1
#define B_   4
#define NROW (B_*S_)
#define NEGV (-1.0e9f)

typedef __attribute__((ext_vector_type(16))) __bf16 v16bf;
typedef __attribute__((ext_vector_type(8)))  __bf16 v8bf;
typedef __attribute__((ext_vector_type(8)))  float  v8f;
typedef __bf16 bf16_t;

// fp32 -> bf16 round-to-nearest-even (scalar)
__device__ __forceinline__ uint32_t f2bf_bits(float f) {
    union { float f; uint32_t u; } c; c.f = f;
    uint32_t u = c.u;
    return (u + 0x7FFFu + ((u >> 16) & 1u)) >> 16;
}
// two fp32 -> packed bf16x2 (one 32-bit LDS store)
__device__ __forceinline__ uint32_t pack2bf(float x, float y) {
    return (f2bf_bits(x) & 0xFFFFu) | (f2bf_bits(y) << 16);
}

// block-wide sum over 256 threads (8 waves, wave32)
__device__ __forceinline__ float blk_reduce_sum(float v, float* sm) {
    int tid = threadIdx.x;
    __syncthreads();
    #pragma unroll
    for (int o = 16; o > 0; o >>= 1) v += __shfl_xor(v, o, 32);
    if ((tid & 31) == 0) sm[tid >> 5] = v;
    __syncthreads();
    if (tid == 0) {
        float t = 0.f;
        #pragma unroll
        for (int i = 0; i < 8; ++i) t += sm[i];
        sm[8] = t;
    }
    __syncthreads();
    return sm[8];
}

// ---------------- embedding + layernorm ----------------
__global__ __launch_bounds__(256) void k_embed(const int* __restrict__ ids,
                                               const float* __restrict__ wemb,
                                               const float* __restrict__ pemb,
                                               const float* __restrict__ gs,
                                               const float* __restrict__ gb,
                                               float* __restrict__ out) {
    int row = blockIdx.x;             // row in [0, NROW)
    int s   = row % S_;
    int tid = threadIdx.x;
    __shared__ float sm[9];
    int id = ids[row];
    float e[3]; float lsum = 0.f;
    #pragma unroll
    for (int i = 0; i < 3; ++i) {
        int d = tid + i * 256;
        e[i] = wemb[(size_t)id * D_ + d] + pemb[(size_t)s * D_ + d];
        lsum += e[i];
    }
    float mean = blk_reduce_sum(lsum, sm) * (1.0f / D_);
    float lv = 0.f;
    #pragma unroll
    for (int i = 0; i < 3; ++i) { float c = e[i] - mean; lv += c * c; }
    float var = blk_reduce_sum(lv, sm) * (1.0f / D_);
    float inv = rsqrtf(var + 1e-5f);
    #pragma unroll
    for (int i = 0; i < 3; ++i) {
        int d = tid + i * 256;
        out[(size_t)row * D_ + d] = (e[i] - mean) * inv * gs[d] + gb[d];
    }
}

// ---------------- layernorm of a buffer ----------------
__global__ __launch_bounds__(256) void k_ln(const float* __restrict__ in,
                                            const float* __restrict__ gs,
                                            const float* __restrict__ gb,
                                            float* __restrict__ out) {
    int row = blockIdx.x;
    int tid = threadIdx.x;
    __shared__ float sm[9];
    float e[3]; float lsum = 0.f;
    #pragma unroll
    for (int i = 0; i < 3; ++i) {
        int d = tid + i * 256;
        e[i] = in[(size_t)row * D_ + d];
        lsum += e[i];
    }
    float mean = blk_reduce_sum(lsum, sm) * (1.0f / D_);
    float lv = 0.f;
    #pragma unroll
    for (int i = 0; i < 3; ++i) { float c = e[i] - mean; lv += c * c; }
    float var = blk_reduce_sum(lv, sm) * (1.0f / D_);
    float inv = rsqrtf(var + 1e-5f);
    #pragma unroll
    for (int i = 0; i < 3; ++i) {
        int d = tid + i * 256;
        out[(size_t)row * D_ + d] = (e[i] - mean) * inv * gs[d] + gb[d];
    }
}

// ---------------- WMMA bf16 GEMM: out = epi(A[MxK] @ W[KxN] + bias) ----------------
// epi: 0 = bias only; 1 = bias + exact GELU; 2 = bias + residual add
//
// LDS tiles stored in WMMA *fragment order* (ISA 7.12.2, 16-bit operands):
//   lane = half*16 + m,  element e -> K = half*8 + e (e<8) | 16 + half*8 + (e-8)
// Fragment fetch = two ds_load_b128 per operand. Staging packs two adjacent-K
// bf16 values into one 32-bit LDS store (adjacent K = same lane, e and e+1).
#define FR_STRIDE 24   // 48-byte per-lane stride (16B aligned, conflict-padded)

__global__ __launch_bounds__(256) void k_gemm(const float* __restrict__ A,
                                              const float* __restrict__ Wt,
                                              const float* __restrict__ bias,
                                              const float* __restrict__ res,
                                              float* __restrict__ out,
                                              int M, int N, int K, int epi) {
    __shared__ __align__(16) bf16_t laf[4][32][FR_STRIDE];
    __shared__ __align__(16) bf16_t lbf[8][32][FR_STRIDE];
    int tid  = threadIdx.x, lane = tid & 31, wid = tid >> 5;
    int wm   = wid & 3;              // 4 waves along M
    int wn   = wid >> 2;             // 2 waves along N
    int row0 = blockIdx.y * 64;
    int col0 = blockIdx.x * 128;
    int m    = lane & 15, half = lane >> 4;

    v8f acc[4];
    #pragma unroll
    for (int i = 0; i < 4; ++i)
        #pragma unroll
        for (int j = 0; j < 8; ++j) acc[i][j] = 0.f;

    // A: 4 K-pairs per thread; pair p = tid + i*256 -> row = p>>4, c0 = (p&15)*2
    // B: 8 K-pairs per thread; pair p = tid + i*256 -> c = p&127, r0 = (p>>7)*2
    float2 rA[4];
    float  rB[16];

    #pragma unroll
    for (int i = 0; i < 4; ++i) {
        int p = tid + i * 256;
        rA[i] = *(const float2*)&A[(size_t)(row0 + (p >> 4)) * K + ((p & 15) << 1)];
    }
    #pragma unroll
    for (int i = 0; i < 8; ++i) {
        int p = tid + i * 256;
        int c = p & 127, r0 = (p >> 7) << 1;
        rB[2 * i]     = Wt[(size_t)(r0)     * N + col0 + c];
        rB[2 * i + 1] = Wt[(size_t)(r0 + 1) * N + col0 + c];
    }

    for (int kk = 0; kk < K; kk += 32) {
        // ---- store current tile into fragment-ordered LDS (packed b32) ----
        #pragma unroll
        for (int i = 0; i < 4; ++i) {
            int p  = tid + i * 256;
            int r  = p >> 4, c0 = (p & 15) << 1;
            int hf = (c0 < 16) ? (c0 >> 3) : ((c0 - 16) >> 3);
            int e0 = (c0 < 16) ? (c0 & 7) : (8 + ((c0 - 16) & 7));
            *(uint32_t*)&laf[r >> 4][hf * 16 + (r & 15)][e0] =
                pack2bf(rA[i].x, rA[i].y);
        }
        #pragma unroll
        for (int i = 0; i < 8; ++i) {
            int p  = tid + i * 256;
            int c  = p & 127, r0 = (p >> 7) << 1;
            int hf = (r0 < 16) ? (r0 >> 3) : ((r0 - 16) >> 3);
            int e0 = (r0 < 16) ? (r0 & 7) : (8 + ((r0 - 16) & 7));
            *(uint32_t*)&lbf[c >> 4][hf * 16 + (c & 15)][e0] =
                pack2bf(rB[2 * i], rB[2 * i + 1]);
        }
        __syncthreads();

        // ---- software pipeline: issue next tile's global loads now ----
        if (kk + 32 < K) {
            #pragma unroll
            for (int i = 0; i < 4; ++i) {
                int p = tid + i * 256;
                rA[i] = *(const float2*)
                    &A[(size_t)(row0 + (p >> 4)) * K + kk + 32 + ((p & 15) << 1)];
            }
            #pragma unroll
            for (int i = 0; i < 8; ++i) {
                int p = tid + i * 256;
                int c = p & 127, r0 = (p >> 7) << 1;
                rB[2 * i]     = Wt[(size_t)(kk + 32 + r0)     * N + col0 + c];
                rB[2 * i + 1] = Wt[(size_t)(kk + 32 + r0 + 1) * N + col0 + c];
            }
            if (kk + 64 < K)
                __builtin_prefetch(&A[(size_t)(row0 + (tid & 63)) * K + kk + 64], 0, 1);
        }

        // ---- fragments via wide LDS loads + 4 WMMAs ----
        v8bf a0 = *(const v8bf*)&laf[wm][lane][0];
        v8bf a1 = *(const v8bf*)&laf[wm][lane][8];
        v16bf af = __builtin_shufflevector(a0, a1,
            0,1,2,3,4,5,6,7,8,9,10,11,12,13,14,15);
        #pragma unroll
        for (int nt = 0; nt < 4; ++nt) {
            v8bf b0 = *(const v8bf*)&lbf[wn * 4 + nt][lane][0];
            v8bf b1 = *(const v8bf*)&lbf[wn * 4 + nt][lane][8];
            v16bf bfm = __builtin_shufflevector(b0, b1,
                0,1,2,3,4,5,6,7,8,9,10,11,12,13,14,15);
            acc[nt] = __builtin_amdgcn_wmma_f32_16x16x32_bf16(
                false, af, false, bfm, (short)0, acc[nt], false, false);
        }
        __syncthreads();
    }

    // epilogue: D layout lane(m,half), vgpr r -> (M = r + 8*half, N = m)
    #pragma unroll
    for (int nt = 0; nt < 4; ++nt) {
        #pragma unroll
        for (int r = 0; r < 8; ++r) {
            int grow = row0 + wm * 16 + r + 8 * half;
            int gcol = col0 + wn * 64 + nt * 16 + m;
            float v = acc[nt][r] + bias[gcol];
            if (epi == 1) {
                v = 0.5f * v * (1.0f + erff(v * 0.70710678118f));
            } else if (epi == 2) {
                v += res[(size_t)grow * N + gcol];
            }
            out[(size_t)grow * N + gcol] = v;
        }
    }
}

// ---------------- sliding-window (band) attention + 1 global key ----------------
// one block per (b, h, chunk c, 16-query sub-block)
__global__ __launch_bounds__(256) void k_band_attn(const float* __restrict__ q,
                                                   const float* __restrict__ k,
                                                   const float* __restrict__ v,
                                                   const int* __restrict__ mask,
                                                   float* __restrict__ out) {
    int bi = blockIdx.x;
    int qb = bi & 7;  bi >>= 3;
    int c  = bi & 15; bi >>= 4;
    int h  = bi % H_;
    int b  = bi / H_;
    int tid = threadIdx.x, lane = tid & 31, wid = tid >> 5;
    __shared__ float qs[16][64];
    __shared__ float sc[16][388];
    int s0 = c * W_ + qb * 16;

    // stage q tile (16x64 fp32 = 4KB) via CDNA5 async global->LDS copy
    {
        int row = tid >> 4, seg = tid & 15;
        const float* gp = q + ((size_t)(b * S_ + s0 + row) * D_ + h * DH_) + seg * 4;
        uint32_t ldsoff = (uint32_t)(uintptr_t)(&qs[row][seg * 4]);
        uint64_t gaddr  = (uint64_t)(uintptr_t)gp;
        asm volatile("global_load_async_to_lds_b128 %0, %1, off"
                     :: "v"(ldsoff), "v"(gaddr) : "memory");
        asm volatile("s_wait_asynccnt 0" ::: "memory");
    }
    __syncthreads();

    const int NC = 3 * W_ + G_;   // 385 score columns: [global | band 0..383]
    for (int col = tid; col < NC; col += 256) {
        bool isg  = (col < G_);
        int  kl   = col - G_;
        int  kpos = isg ? 0 : ((c - 1) * W_ + kl);
        int  kpc  = min(max(kpos, 0), S_ - 1);
        bool inb  = isg || (kpos >= 0 && kpos < S_);
        bool km   = mask[b * S_ + kpc] > 0;
        const float* kr = k + ((size_t)(b * S_ + kpc) * D_ + h * DH_);
        #pragma unroll 1
        for (int r = 0; r < 16; ++r) {
            int ql = qb * 16 + r;
            int dd = kl - W_ - ql;
            bool band = isg || (dd >= -W_ && dd <= W_);
            float a = 0.f;
            #pragma unroll 8
            for (int d = 0; d < DH_; ++d) a += qs[r][d] * kr[d];
            sc[r][col] = (inb && km && band) ? a * 0.125f : NEGV;
        }
    }
    __syncthreads();

    // softmax: each wave handles 2 rows
    for (int r = wid * 2; r < wid * 2 + 2; ++r) {
        float mx = -3.4e38f;
        for (int col = lane; col < NC; col += 32) mx = fmaxf(mx, sc[r][col]);
        #pragma unroll
        for (int o = 16; o > 0; o >>= 1) mx = fmaxf(mx, __shfl_xor(mx, o, 32));
        float sum = 0.f;
        for (int col = lane; col < NC; col += 32) {
            float e = __expf(sc[r][col] - mx);
            sc[r][col] = e; sum += e;
        }
        #pragma unroll
        for (int o = 16; o > 0; o >>= 1) sum += __shfl_xor(sum, o, 32);
        float inv = 1.0f / sum;
        for (int col = lane; col < NC; col += 32) sc[r][col] *= inv;
    }
    __syncthreads();

    // out = probs @ [vg | v3]
    #pragma unroll 1
    for (int p = 0; p < 4; ++p) {
        int idx = tid + p * 256;
        int r = idx >> 6, d = idx & 63;
        float acc = 0.f;
        for (int col = 0; col < NC; ++col) {
            int kpos = (col < G_) ? 0 : ((c - 1) * W_ + (col - G_));
            int kpc  = min(max(kpos, 0), S_ - 1);
            acc += sc[r][col] * v[(size_t)(b * S_ + kpc) * D_ + h * DH_ + d];
        }
        out[(size_t)(b * S_ + s0 + r) * D_ + h * DH_ + d] = acc;
    }
}

// ---------------- full attention for the single global token (s==0) ----------------
__global__ __launch_bounds__(256) void k_global_attn(const float* __restrict__ q,
                                                     const float* __restrict__ k,
                                                     const float* __restrict__ v,
                                                     const int* __restrict__ mask,
                                                     float* __restrict__ out) {
    int h = blockIdx.x % H_, b = blockIdx.x / H_;
    int tid = threadIdx.x;
    __shared__ float sc[S_];
    __shared__ float qv[DH_];
    __shared__ float red[16];
    __shared__ float po[4][64];
    if (tid < DH_) qv[tid] = q[(size_t)(b * S_) * D_ + h * DH_ + tid] * 0.125f;
    __syncthreads();
    float lmax = -3.4e38f;
    for (int s = tid; s < S_; s += 256) {
        const float* kr = k + ((size_t)(b * S_ + s) * D_ + h * DH_);
        float a = 0.f;
        #pragma unroll 8
        for (int d = 0; d < DH_; ++d) a += qv[d] * kr[d];
        a = (mask[b * S_ + s] > 0) ? a : NEGV;
        sc[s] = a; lmax = fmaxf(lmax, a);
    }
    #pragma unroll
    for (int o = 16; o > 0; o >>= 1) lmax = fmaxf(lmax, __shfl_xor(lmax, o, 32));
    if ((tid & 31) == 0) red[tid >> 5] = lmax;
    __syncthreads();
    if (tid == 0) { float m0 = red[0]; for (int i = 1; i < 8; ++i) m0 = fmaxf(m0, red[i]); red[8] = m0; }
    __syncthreads();
    float mx = red[8];
    float lsum = 0.f;
    for (int s = tid; s < S_; s += 256) { float e = __expf(sc[s] - mx); sc[s] = e; lsum += e; }
    #pragma unroll
    for (int o = 16; o > 0; o >>= 1) lsum += __shfl_xor(lsum, o, 32);
    if ((tid & 31) == 0) red[tid >> 5] = lsum;
    __syncthreads();
    if (tid == 0) { float t = 0.f; for (int i = 0; i < 8; ++i) t += red[i]; red[9] = t; }
    __syncthreads();
    float inv = 1.0f / red[9];
    int d = tid & 63, grp = tid >> 6;
    float acc = 0.f;
    for (int s = grp * 512; s < grp * 512 + 512; ++s)
        acc += sc[s] * v[(size_t)(b * S_ + s) * D_ + h * DH_ + d];
    po[grp][d] = acc;
    __syncthreads();
    if (tid < 64)
        out[(size_t)(b * S_) * D_ + h * DH_ + tid] =
            (po[0][tid] + po[1][tid] + po[2][tid] + po[3][tid]) * inv;
}

// ---------------- classifier head: logits[b] = x[b,0,:] . Wc + bc ----------------
__global__ __launch_bounds__(256) void k_cls(const float* __restrict__ x,
                                             const float* __restrict__ Wc,
                                             const float* __restrict__ bc,
                                             float* __restrict__ out) {
    int b = blockIdx.x, tid = threadIdx.x;
    __shared__ float sm[9];
    float l = 0.f;
    #pragma unroll
    for (int i = 0; i < 3; ++i) {
        int d = tid + i * 256;
        l += x[(size_t)(b * S_) * D_ + d] * Wc[d];
    }
    float t = blk_reduce_sum(l, sm);
    if (tid == 0) out[b] = t + bc[0];
}

// ---------------- host launcher ----------------
extern "C" void kernel_launch(void* const* d_in, const int* in_sizes, int n_in,
                              void* d_out, int out_size, void* d_ws, size_t ws_size,
                              hipStream_t stream) {
    (void)in_sizes; (void)n_in; (void)out_size; (void)ws_size;
    const float* wemb = (const float*)d_in[0];
    const float* pemb = (const float*)d_in[1];
    const float* lnes = (const float*)d_in[2];
    const float* lneb = (const float*)d_in[3];
    const float* Wq   = (const float*)d_in[4];
    const float* bq   = (const float*)d_in[5];
    const float* Wk   = (const float*)d_in[6];
    const float* bk   = (const float*)d_in[7];
    const float* Wv   = (const float*)d_in[8];
    const float* bv   = (const float*)d_in[9];
    const float* Wo   = (const float*)d_in[10];
    const float* bo   = (const float*)d_in[11];
    const float* ln1s = (const float*)d_in[12];
    const float* ln1b = (const float*)d_in[13];
    const float* W1   = (const float*)d_in[14];
    const float* b1   = (const float*)d_in[15];
    const float* W2   = (const float*)d_in[16];
    const float* b2   = (const float*)d_in[17];
    const float* ln2s = (const float*)d_in[18];
    const float* ln2b = (const float*)d_in[19];
    const float* Wc   = (const float*)d_in[20];
    const float* bc   = (const float*)d_in[21];
    const int*   ids  = (const int*)d_in[22];
    const int*   mask = (const int*)d_in[23];
    float* out = (float*)d_out;

    // workspace layout (all fp32)
    size_t nd = (size_t)NROW * D_;
    float* ax = (float*)d_ws;        // residual / layer input  [NROW, D]
    float* aq = ax + nd;             // Q                        [NROW, D]
    float* ak = aq + nd;             // K
    float* av = ak + nd;             // V
    float* aa = av + nd;             // attention output
    float* at = aa + nd;             // temp (pre-LN)
    float* ah = at + nd;             // FFN hidden               [NROW, FF]

    k_embed<<<NROW, 256, 0, stream>>>(ids, wemb, pemb, lnes, lneb, ax);

    dim3 gD(D_ / 128, NROW / 64);    // 6 x 128 blocks
    dim3 gF(FF_ / 128, NROW / 64);   // 24 x 128 blocks

    for (int l = 0; l < L_; ++l) {
        const float* wq = Wq + (size_t)l * D_ * D_;
        const float* wk = Wk + (size_t)l * D_ * D_;
        const float* wv = Wv + (size_t)l * D_ * D_;
        const float* wo = Wo + (size_t)l * D_ * D_;
        const float* w1 = W1 + (size_t)l * D_ * FF_;
        const float* w2 = W2 + (size_t)l * FF_ * D_;

        k_gemm<<<gD, 256, 0, stream>>>(ax, wq, bq + (size_t)l * D_, nullptr, aq, NROW, D_, D_, 0);
        k_gemm<<<gD, 256, 0, stream>>>(ax, wk, bk + (size_t)l * D_, nullptr, ak, NROW, D_, D_, 0);
        k_gemm<<<gD, 256, 0, stream>>>(ax, wv, bv + (size_t)l * D_, nullptr, av, NROW, D_, D_, 0);

        k_band_attn<<<B_ * H_ * C_ * 8, 256, 0, stream>>>(aq, ak, av, mask, aa);
        k_global_attn<<<B_ * H_, 256, 0, stream>>>(aq, ak, av, mask, aa);

        k_gemm<<<gD, 256, 0, stream>>>(aa, wo, bo + (size_t)l * D_, ax, at, NROW, D_, D_, 2);
        k_ln<<<NROW, 256, 0, stream>>>(at, ln1s + (size_t)l * D_, ln1b + (size_t)l * D_, ax);

        k_gemm<<<gF, 256, 0, stream>>>(ax, w1, b1 + (size_t)l * FF_, nullptr, ah, NROW, FF_, D_, 1);
        k_gemm<<<gD, 256, 0, stream>>>(ah, w2, b2 + (size_t)l * D_, ax, at, NROW, D_, FF_, 2);
        k_ln<<<NROW, 256, 0, stream>>>(at, ln2s + (size_t)l * D_, ln2b + (size_t)l * D_, ax);
    }

    k_cls<<<B_, 256, 0, stream>>>(ax, Wc, bc, out);
}
